// QKNormedMultiHeadAttention_55800215109734
// MI455X (gfx1250) — matrix operand order repaired and checked by reading, not compiled, not run on hardware
//
#include <hip/hip_runtime.h>
#include <math.h>

typedef __bf16 bf16;
typedef __attribute__((ext_vector_type(16))) __bf16 v16bf;
typedef __attribute__((ext_vector_type(8)))  __bf16 v8bf;
typedef __attribute__((ext_vector_type(4)))  __bf16 v4bf;
typedef __attribute__((ext_vector_type(8)))  float  v8f;

#define B_DIM 2
#define S_DIM 2048
#define D_DIM 1024
#define H_DIM 16
#define DH    64
#define EPSF  1e-8f

__device__ __forceinline__ v16bf make_v16(v8bf lo, v8bf hi) {
  return __builtin_shufflevector(lo, hi, 0,1,2,3,4,5,6,7,8,9,10,11,12,13,14,15);
}

// Low 32 bits of a generic LDS address = byte offset within the workgroup's
// LDS allocation.
__device__ __forceinline__ unsigned lds_off(const void* p) {
  return (unsigned)(size_t)p;
}

// Async global->LDS copy of 16 bytes per lane (gfx1250 VGLOBAL async op,
// tracked by ASYNCcnt).
__device__ __forceinline__ void async_copy_b128(const void* gaddr, unsigned ldsByteOff) {
  asm volatile("global_load_async_to_lds_b128 %0, %1, off"
               :: "v"(ldsByteOff), "v"(gaddr) : "memory");
}
__device__ __forceinline__ void wait_async0() {
  asm volatile("s_wait_asynccnt 0" ::: "memory");
}

// ---------------------------------------------------------------- convert ---
__global__ __launch_bounds__(256) void cvt_f32_bf16(const float* __restrict__ in,
                                                    bf16* __restrict__ out, int n4) {
  int i = blockIdx.x * blockDim.x + threadIdx.x;
  if (i < n4) {
    float4 f = ((const float4*)in)[i];
    v4bf o; o[0]=(bf16)f.x; o[1]=(bf16)f.y; o[2]=(bf16)f.z; o[3]=(bf16)f.w;
    *(v4bf*)(out + (size_t)i*4) = o;
  }
}

// Wt[n*K + k] = (bf16)W[k*N + n] : one-time weight transpose so GEMM B-tiles
// are contiguous (enables async-to-LDS staging, kills the LDS scatter).
__global__ __launch_bounds__(256) void cvt_transpose_bf16(const float* __restrict__ in,
                                                          bf16* __restrict__ out,
                                                          int K, int N) {
  int idx = blockIdx.x * blockDim.x + threadIdx.x;   // over N*K, k fastest
  if (idx < N * K) {
    int n = idx / K, k = idx - n * K;
    out[idx] = (bf16)in[(size_t)k * N + n];
  }
}

// ------------------------------------------------------------------- GEMM ---
// C[M,N] f32 = A[M,K] bf16 @ Bt[N,K] bf16 (B pre-transposed).
// Block: 256 thr (8 waves, 2x4), macro tile 64x128, K-step 32.
// Both LDS tiles staged with global_load_async_to_lds_b128.
__global__ __launch_bounds__(256) void gemm_bf16_f32(const bf16* __restrict__ A,
                                                     const bf16* __restrict__ Bt,
                                                     float* __restrict__ C,
                                                     int M, int N, int K) {
  __shared__ bf16 As[64 * 32];     // [row][k]
  __shared__ bf16 Bs[128 * 32];    // [n][k]
  const int tid  = threadIdx.x;
  const int lane = tid & 31;
  const int wv   = tid >> 5;
  const int wm   = wv & 1;         // M 32-half
  const int wn   = wv >> 1;        // N 32-quarter
  const int m0   = blockIdx.y * 64;
  const int n0   = blockIdx.x * 128;
  const int half = lane >> 4;
  const int l16  = lane & 15;

  v8f acc[2][2] = {};

  const int arow = tid >> 2;             // 0..63
  const int acol = (tid & 3) * 8;        // 0,8,16,24
  const int brow = tid & 127;            // 0..127 (n within tile)
  const int bkc  = (tid >> 7) * 16;      // 0 or 16

  const bf16* aSrc = A  + (size_t)(m0 + arow) * K + acol;
  const bf16* bSrc = Bt + (size_t)(n0 + brow) * K + bkc;
  const unsigned aDst  = lds_off(&As[arow * 32 + acol]);
  const unsigned bDst0 = lds_off(&Bs[brow * 32 + bkc]);

  for (int k0 = 0; k0 < K; k0 += 32) {
    __syncthreads();                       // WAR: prior iter's frag reads done
    async_copy_b128(aSrc + k0, aDst);
    async_copy_b128(bSrc + k0,     bDst0);
    async_copy_b128(bSrc + k0 + 8, bDst0 + 16);
    if (k0 + 32 < K) {
      __builtin_prefetch(aSrc + k0 + 32, 0, 0);
      __builtin_prefetch(bSrc + k0 + 32, 0, 0);
    }
    wait_async0();
    __syncthreads();

    // A fragments: lanes 0-15 hold K{0..7,16..23}, lanes 16-31 K{8..15,24..31}
    v16bf afrag[2], bfrag[2];
#pragma unroll
    for (int fi = 0; fi < 2; fi++) {
      const bf16* ap = &As[(wm * 32 + fi * 16 + l16) * 32 + half * 8];
      afrag[fi] = make_v16(*(const v8bf*)(ap), *(const v8bf*)(ap + 16));
    }
    // B fragments: lanes 0-15 hold K{0..7,16..23}, lanes 16-31 K{8..15,24..31}
#pragma unroll
    for (int fj = 0; fj < 2; fj++) {
      const bf16* bp = &Bs[(wn * 32 + fj * 16 + l16) * 32 + half * 8];
      bfrag[fj] = make_v16(*(const v8bf*)(bp), *(const v8bf*)(bp + 16));
    }
#pragma unroll
    for (int fi = 0; fi < 2; fi++)
#pragma unroll
      for (int fj = 0; fj < 2; fj++)
        acc[fi][fj] = __builtin_amdgcn_wmma_f32_16x16x32_bf16(
            false, afrag[fi], false, bfrag[fj], (short)0, acc[fi][fj], false, false);
  }

#pragma unroll
  for (int fi = 0; fi < 2; fi++)
#pragma unroll
    for (int fj = 0; fj < 2; fj++) {
      int n = n0 + wn * 32 + fj * 16 + l16;
#pragma unroll
      for (int v = 0; v < 8; v++) {
        int m = m0 + wm * 32 + fi * 16 + v + half * 8;
        C[(size_t)m * N + n] = acc[fi][fj][v];
      }
    }
}

// ------------------------------------------------------- normalize + split ---
// qkvf [B,S,3D] f32 -> Qb/Kb [B,H,S,DH] bf16 (Q pre-scaled tau/(norm*sqrt(DH))),
// Vt [B,H,DH,S] bf16 (transposed for contiguous P@V B-fragments).
__global__ __launch_bounds__(64) void qkv_norm_split(const float* __restrict__ qkvf,
                                                     const float* __restrict__ tau,
                                                     bf16* __restrict__ Qb,
                                                     bf16* __restrict__ Kb,
                                                     bf16* __restrict__ Vt) {
  const int t = threadIdx.x;            // dh
  const int s = blockIdx.x, h = blockIdx.y, b = blockIdx.z;
  const float* row = qkvf + ((size_t)(b * S_DIM + s)) * (3 * D_DIM) + h * DH;
  float q = row[t], k = row[D_DIM + t], v = row[2 * D_DIM + t];
  float sq = q * q, sk = k * k;
#pragma unroll
  for (int sft = 1; sft < 32; sft <<= 1) {
    sq += __shfl_xor(sq, sft, 32);
    sk += __shfl_xor(sk, sft, 32);
  }
  __shared__ float red[4];
  if ((t & 31) == 0) { red[t >> 5] = sq; red[2 + (t >> 5)] = sk; }
  __syncthreads();
  float nq = red[0] + red[1];
  float nk = red[2] + red[3];
  float qscale = tau[h] / ((sqrtf(nq) + EPSF) * 8.0f);   // tau/(norm)/sqrt(DH)
  float kscale = 1.0f / (sqrtf(nk) + EPSF);
  size_t bh = (size_t)(b * H_DIM + h);
  Qb[(bh * S_DIM + s) * DH + t] = (bf16)(q * qscale);
  Kb[(bh * S_DIM + s) * DH + t] = (bf16)(k * kscale);
  Vt[(bh * DH + t) * S_DIM + s] = (bf16)v;
}

// -------------------------------------------------------- flash attention ---
// 8 independent waves per block; each wave owns 16 queries. Online softmax
// over 32-key blocks, causal. QK^T and P@V via bf16 WMMA (f32 accum).
#define FA_WAVES 8
__global__ __launch_bounds__(256) void flash_attn(const bf16* __restrict__ Qb,
                                                  const bf16* __restrict__ Kb,
                                                  const bf16* __restrict__ Vt,
                                                  bf16* __restrict__ Attn) {
  __shared__ bf16 Pld[FA_WAVES][16 * 32];
  const int tid  = threadIdx.x;
  const int lane = tid & 31;
  const int wv   = tid >> 5;
  const int half = lane >> 4;
  const int l16  = lane & 15;
  const int b = blockIdx.z, h = blockIdx.y;
  const int q0 = blockIdx.x * (16 * FA_WAVES) + wv * 16;

  const size_t bh = (size_t)(b * H_DIM + h);
  const bf16* Qp = Qb + bh * S_DIM * DH;
  const bf16* Kp = Kb + bh * S_DIM * DH;
  const bf16* Vp = Vt + bh * DH * S_DIM;

  // Q A-fragments for two K-steps of DH
  v16bf qf[2];
  {
    const bf16* qrow = Qp + (size_t)(q0 + l16) * DH;
#pragma unroll
    for (int ks = 0; ks < 2; ks++) {
      int off = ks * 32 + half * 8;
      qf[ks] = make_v16(*(const v8bf*)(qrow + off), *(const v8bf*)(qrow + off + 16));
    }
  }

  v8f o[4] = {};
  float rmax[8], rsum[8];
#pragma unroll
  for (int v = 0; v < 8; v++) { rmax[v] = -3.0e30f; rsum[v] = 0.f; }

  bf16* myP = Pld[wv];

  for (int kv0 = 0; kv0 <= q0 + 15; kv0 += 32) {
    if (kv0 + 32 <= q0 + 15) {
      __builtin_prefetch(Kp + (size_t)(kv0 + 32 + l16) * DH, 0, 0);
      __builtin_prefetch(Vp + (size_t)(l16) * S_DIM + kv0 + 32, 0, 0);
    }
    // S = Q K^T : two 16-key n-tiles, two K-steps each
    v8f sf[2] = {};
#pragma unroll
    for (int nt = 0; nt < 2; nt++) {
#pragma unroll
      for (int ks = 0; ks < 2; ks++) {
        const bf16* krow = Kp + (size_t)(kv0 + nt * 16 + l16) * DH + ks * 32 + half * 8;
        v16bf kf = make_v16(*(const v8bf*)(krow), *(const v8bf*)(krow + 16));
        sf[nt] = __builtin_amdgcn_wmma_f32_16x16x32_bf16(
            false, qf[ks], false, kf, (short)0, sf[nt], false, false);
      }
    }
    // causal mask + online softmax (rows m = v + 8*half; cols across 16 lanes)
#pragma unroll
    for (int v = 0; v < 8; v++) {
      int qg = q0 + v + half * 8;
#pragma unroll
      for (int nt = 0; nt < 2; nt++) {
        int kg = kv0 + nt * 16 + l16;
        if (kg > qg) sf[nt][v] = -3.0e30f;
      }
      float mv = fmaxf(sf[0][v], sf[1][v]);
#pragma unroll
      for (int s = 1; s < 16; s <<= 1) mv = fmaxf(mv, __shfl_xor(mv, s, 32));
      float mnew = fmaxf(rmax[v], mv);
      float corr = __expf(rmax[v] - mnew);
      rmax[v] = mnew;
      float p0 = __expf(sf[0][v] - mnew);
      float p1 = __expf(sf[1][v] - mnew);
      sf[0][v] = p0; sf[1][v] = p1;
      float ps = p0 + p1;
#pragma unroll
      for (int s = 1; s < 16; s <<= 1) ps += __shfl_xor(ps, s, 32);
      rsum[v] = rsum[v] * corr + ps;
#pragma unroll
      for (int nt = 0; nt < 4; nt++) o[nt][v] *= corr;
    }
    // C-layout -> A-layout transpose of P through wave-private LDS
#pragma unroll
    for (int v = 0; v < 8; v++) {
      int m = v + half * 8;
      myP[m * 32 + l16]      = (bf16)sf[0][v];
      myP[m * 32 + 16 + l16] = (bf16)sf[1][v];
    }
    asm volatile("s_wait_dscnt 0" ::: "memory");
    v16bf pf;
    {
      const bf16* prow = myP + l16 * 32 + half * 8;
      pf = make_v16(*(const v8bf*)(prow), *(const v8bf*)(prow + 16));
    }
    // O += P @ V : four dh n-tiles, Vt gives B-fragments as two contiguous 8-key runs
#pragma unroll
    for (int nt = 0; nt < 4; nt++) {
      const bf16* vrow = Vp + (size_t)(nt * 16 + l16) * S_DIM + kv0 + half * 8;
      v16bf vf = make_v16(*(const v8bf*)(vrow), *(const v8bf*)(vrow + 16));
      o[nt] = __builtin_amdgcn_wmma_f32_16x16x32_bf16(
          false, pf, false, vf, (short)0, o[nt], false, false);
    }
  }

  // normalize rows, emit attn as [B,S,H*DH] bf16 for the output GEMM
#pragma unroll
  for (int v = 0; v < 8; v++) {
    float inv = 1.0f / rsum[v];
    int qg = q0 + v + half * 8;
    bf16* orow = Attn + ((size_t)b * S_DIM + qg) * D_DIM + h * DH;
#pragma unroll
    for (int nt = 0; nt < 4; nt++)
      orow[nt * 16 + l16] = (bf16)(o[nt][v] * inv);
  }
}

// ----------------------------------------------------------------- launch ---
extern "C" void kernel_launch(void* const* d_in, const int* in_sizes, int n_in,
                              void* d_out, int out_size, void* d_ws, size_t ws_size,
                              hipStream_t stream) {
  const float* x    = (const float*)d_in[0];
  // d_in[1] = mask [B,S,S] bool: causal tril; masking implemented analytically.
  const float* Wqkv = (const float*)d_in[2];
  const float* Wo   = (const float*)d_in[3];
  const float* tau  = (const float*)d_in[4];
  float* out = (float*)d_out;

  const size_t M   = (size_t)B_DIM * S_DIM;     // 4096
  const size_t N1  = 3 * D_DIM;                 // 3072
  const size_t KD  = D_DIM;                     // 1024
  const size_t HSD = (size_t)B_DIM * H_DIM * S_DIM * DH;  // 4M elems

  char* w = (char*)d_ws;
  bf16*  xb     = (bf16*)w;  w += M * KD * sizeof(bf16);        // 8 MB
  bf16*  wqkvT  = (bf16*)w;  w += KD * N1 * sizeof(bf16);       // 6 MB  [N1,K]
  bf16*  woT    = (bf16*)w;  w += KD * KD * sizeof(bf16);       // 2 MB  [K,K]
  float* qkvf   = (float*)w; w += M * N1 * sizeof(float);       // 48 MB
  bf16*  Qb     = (bf16*)w;  w += HSD * sizeof(bf16);           // 8 MB
  bf16*  Kb     = (bf16*)w;  w += HSD * sizeof(bf16);           // 8 MB
  bf16*  Vt     = (bf16*)w;  w += HSD * sizeof(bf16);           // 8 MB
  bf16*  attnb  = (bf16*)qkvf;  // reuse qkvf region (consumed before attn write)

  // 1) operand conversion (x straight; weights transposed to [N,K] bf16)
  {
    int n4 = (int)(M * KD / 4);
    cvt_f32_bf16<<<(n4 + 255) / 256, 256, 0, stream>>>(x, xb, n4);
    int nt1 = (int)(KD * N1);
    cvt_transpose_bf16<<<(nt1 + 255) / 256, 256, 0, stream>>>(Wqkv, wqkvT, (int)KD, (int)N1);
    int nt2 = (int)(KD * KD);
    cvt_transpose_bf16<<<(nt2 + 255) / 256, 256, 0, stream>>>(Wo, woT, (int)KD, (int)KD);
  }
  // 2) QKV projection (WMMA, async-to-LDS staging)
  gemm_bf16_f32<<<dim3((unsigned)(N1 / 128), (unsigned)(M / 64)), 256, 0, stream>>>(
      xb, wqkvT, qkvf, (int)M, (int)N1, (int)KD);
  // 3) L2-norm Q/K, fold tau/sqrt(DH), split + V transpose
  qkv_norm_split<<<dim3(S_DIM, H_DIM, B_DIM), 64, 0, stream>>>(qkvf, tau, Qb, Kb, Vt);
  // 4) causal flash attention (WMMA)
  flash_attn<<<dim3(S_DIM / (16 * FA_WAVES), H_DIM, B_DIM), 256, 0, stream>>>(
      Qb, Kb, Vt, attnb);
  // 5) output projection (WMMA) straight into d_out
  gemm_bf16_f32<<<dim3((unsigned)(KD / 128), (unsigned)(M / 64)), 256, 0, stream>>>(
      attnb, woT, out, (int)M, (int)KD, (int)KD);
}
